// GATConv_14173392077050
// MI455X (gfx1250) — compile-verified, matrix operand order checked by reading, not run on hardware
//
#include <hip/hip_runtime.h>

// ---------------------------------------------------------------------------
// GATConv for MI455X (gfx1250, wave32).
//   N=100000 nodes, E=1600000 edges, D_IN=256, H=4 heads x F=64 = 256 out.
// Dense projections: async-copy feat tile to LDS (ASYNCcnt), convert to bf16
// A-fragments in LDS once per block, then v_wmma_f32_16x16x32_bf16 with
// B-fragments pre-packed in L2. Edge softmax via sortable-uint atomicMax +
// f32 atomicAdd; aggregation exploits sorted dst with register accumulation
// + atomic flush (gathers are L2-resident: h_src is 102MB < 192MB L2).
// ---------------------------------------------------------------------------

typedef __attribute__((ext_vector_type(16))) __bf16 v16bf;
typedef __attribute__((ext_vector_type(8)))  float  v8f;

#define N_NODES 100000
#define N_EDGES 1600000
#define D_IN    256          // == H*F
#define NHEAD   4

// workspace byte offsets (all multiples of 32 for v16bf loads)
#define OFF_HSRC   ((size_t)0)            // 100000*256*4  = 102,400,000
#define OFF_LOGIT  ((size_t)102400000)    // 1600000*4*4   =  25,600,000
#define OFF_EMAX   ((size_t)128000000)    // 100000*4*4    =   1,600,000
#define OFF_DENOM  ((size_t)129600000)    // 100000*4*4    =   1,600,000
#define OFF_WPACK  ((size_t)131200000)    // 2*8*16*32*16*2=     262,144

// ---- sortable-uint encoding of float (order-preserving) -------------------
__device__ __forceinline__ unsigned enc_f(float f) {
  unsigned u = __float_as_uint(f);
  return (u & 0x80000000u) ? ~u : (u | 0x80000000u);
}
__device__ __forceinline__ float dec_f(unsigned u) {
  u = (u & 0x80000000u) ? (u & 0x7FFFFFFFu) : ~u;
  return __uint_as_float(u);
}

// ---------------------------------------------------------------------------
// Kernel 1: pack W_src / W_dst (fp32, [K=256][N=256] row-major) into bf16
// WMMA B-fragments. Fragment layout per CDNA5 ISA (16-bit 32x16 B):
//   lane l: N = (l&15)+16*ct ; kb = (l<16)?0:8
//   element e: K = 32*kt + kb + (e<8 ? e : e+8)
// One thread per (w, kt, ct, lane) writes 16 contiguous bf16 (32 bytes).
// Packed W = 256 KB total -> stays L2-resident for all 6250 row-tile blocks.
// ---------------------------------------------------------------------------
__global__ __launch_bounds__(256) void pack_w_kernel(
    const float* __restrict__ Ws, const float* __restrict__ Wd,
    __bf16* __restrict__ wpack) {
  int t = blockIdx.x * blockDim.x + threadIdx.x;
  if (t >= 2 * 8 * 16 * 32) return;
  int lane = t & 31;
  int ct   = (t >> 5) & 15;
  int kt   = (t >> 9) & 7;
  int w    = t >> 12;
  const float* W = w ? Wd : Ws;
  int n  = (lane & 15) + ct * 16;
  int kb = (lane < 16) ? 0 : 8;
  __bf16* dp = wpack + (size_t)t * 16;
#pragma unroll
  for (int e = 0; e < 16; ++e) {
    int k = kt * 32 + kb + ((e < 8) ? e : e + 8);
    dp[e] = (__bf16)W[(size_t)k * D_IN + n];
  }
}

// ---------------------------------------------------------------------------
// Kernel 2: per-call init of segment-max (enc(-inf)) and denominators (0)
// ---------------------------------------------------------------------------
__global__ __launch_bounds__(256) void init_kernel(
    unsigned* __restrict__ emax, float* __restrict__ denom, int n) {
  int i = blockIdx.x * blockDim.x + threadIdx.x;
  if (i < n) {
    emax[i]  = enc_f(-__builtin_inff());
    denom[i] = 0.0f;
  }
}

// ---------------------------------------------------------------------------
// Kernel 3: WMMA GEMM. Block = 128 thr = 4 waves, one 16-row tile per block.
// Stage: (a) async-copy 16x256 fp32 feat tile to LDS (ASYNCcnt path),
//        (b) convert to bf16 A-fragments in LDS (once per block),
//        (c) each wave computes 16x64 cols of BOTH h_src (->ws) and
//            h_dst+bias (->d_out residual): 8 WMMA accumulators per wave.
// ---------------------------------------------------------------------------
__global__ __launch_bounds__(128) void gemm_kernel(
    const float* __restrict__ feat, const __bf16* __restrict__ wpack,
    const float* __restrict__ bias, float* __restrict__ hsrc,
    float* __restrict__ outdst) {
  __shared__ __align__(16) float  sfeat[16 * D_IN];       // 16 KB fp32 tile
  __shared__ __align__(32) __bf16 safrag[8 * 32 * 16];    // 8 KB A-fragments

  const int tid  = threadIdx.x;
  const int wave = tid >> 5;
  const int lane = tid & 31;
  const int row0 = blockIdx.x * 16;

  // ---- (a) async global->LDS copy of the fp32 feat tile ----
  {
    unsigned long long g0 =
        (unsigned long long)(const void*)(feat + (size_t)row0 * D_IN);
    unsigned l0 = (unsigned)(unsigned long long)(const void*)(&sfeat[0]);
#pragma unroll
    for (int r = 0; r < 8; ++r) {
      int i = r * 128 + tid;  // which 16-byte packet of the 16 KB tile
      unsigned long long g = g0 + (unsigned long long)i * 16u;
      unsigned           l = l0 + (unsigned)i * 16u;
      asm volatile("global_load_async_to_lds_b128 %0, %1, off"
                   :: "v"(l), "v"(g) : "memory");
    }
    asm volatile("s_wait_asynccnt 0" ::: "memory");
  }
  __syncthreads();

  // ---- (b) fp32 -> bf16 A-fragment conversion in LDS (ISA 7.12.2 layout) --
#pragma unroll
  for (int q = 0; q < 2; ++q) {
    int idx = q * 128 + tid;       // (kt, lane) in [0, 256)
    int kt  = idx >> 5;
    int l   = idx & 31;
    int m   = l & 15;
    int kb  = (l < 16) ? 0 : 8;
    const float* sp = &sfeat[m * D_IN + kt * 32 + kb];
    float4 f0 = *(const float4*)(sp);
    float4 f1 = *(const float4*)(sp + 4);
    float4 f2 = *(const float4*)(sp + 16);
    float4 f3 = *(const float4*)(sp + 20);
    v16bf a;
    a[0]  = (__bf16)f0.x; a[1]  = (__bf16)f0.y; a[2]  = (__bf16)f0.z; a[3]  = (__bf16)f0.w;
    a[4]  = (__bf16)f1.x; a[5]  = (__bf16)f1.y; a[6]  = (__bf16)f1.z; a[7]  = (__bf16)f1.w;
    a[8]  = (__bf16)f2.x; a[9]  = (__bf16)f2.y; a[10] = (__bf16)f2.z; a[11] = (__bf16)f2.w;
    a[12] = (__bf16)f3.x; a[13] = (__bf16)f3.y; a[14] = (__bf16)f3.z; a[15] = (__bf16)f3.w;
    *(v16bf*)(&safrag[(size_t)idx * 16]) = a;
  }
  __syncthreads();

  // ---- (c) WMMA main loop: 8 K-steps, 8 output tiles per wave -------------
  v8f acc[8] = {};
  for (int kt = 0; kt < 8; ++kt) {
    v16bf a = *(const v16bf*)(&safrag[(size_t)(kt * 32 + lane) * 16]);
#pragma unroll
    for (int w = 0; w < 2; ++w) {
      const __bf16* wb =
          wpack + ((((size_t)w * 8 + kt) * 16 + wave * 4) * 32 + lane) * 16;
#pragma unroll
      for (int t = 0; t < 4; ++t) {
        v16bf b = *(const v16bf*)(wb + (size_t)t * 32 * 16);
        acc[w * 4 + t] = __builtin_amdgcn_wmma_f32_16x16x32_bf16(
            false, a, false, b, (short)0, acc[w * 4 + t], false, false);
      }
    }
  }

  // ---- store: C-fragment layout (VGPR r: lanes 0-15 -> M=r, 16-31 -> M=r+8)
#pragma unroll
  for (int w = 0; w < 2; ++w) {
    float* op = w ? outdst : hsrc;
#pragma unroll
    for (int t = 0; t < 4; ++t) {
      int col = wave * 64 + t * 16 + (lane & 15);
      float bv = w ? bias[col] : 0.0f;
#pragma unroll
      for (int r = 0; r < 8; ++r) {
        int rowM = (lane < 16) ? r : r + 8;
        op[(size_t)(row0 + rowM) * D_IN + col] = acc[w * 4 + t][r] + bv;
      }
    }
  }
}

// ---------------------------------------------------------------------------
// Kernel 4: attention projections a_src = feat@Wa_src, a_dst = feat@Wa_dst.
// One wave per node; lane owns 8 of 256 k's; shfl_xor tree reduction.
// ---------------------------------------------------------------------------
__global__ __launch_bounds__(256) void attn_proj_kernel(
    const float* __restrict__ feat, const float* __restrict__ WaS,
    const float* __restrict__ WaD, float* __restrict__ aS,
    float* __restrict__ aD, int n) {
  int node = blockIdx.x * 8 + (threadIdx.x >> 5);
  int lane = threadIdx.x & 31;
  if (node >= n) return;
  const float* fr = feat + (size_t)node * D_IN + lane * 8;
  float4 f0 = *(const float4*)(fr);
  float4 f1 = *(const float4*)(fr + 4);
  float fv[8] = {f0.x, f0.y, f0.z, f0.w, f1.x, f1.y, f1.z, f1.w};
  float ps[NHEAD] = {0, 0, 0, 0}, pd[NHEAD] = {0, 0, 0, 0};
#pragma unroll
  for (int j = 0; j < 8; ++j) {
    int k = lane * 8 + j;
    float4 ws = *(const float4*)(WaS + (size_t)k * NHEAD);
    float4 wd = *(const float4*)(WaD + (size_t)k * NHEAD);
    ps[0] += fv[j] * ws.x; ps[1] += fv[j] * ws.y;
    ps[2] += fv[j] * ws.z; ps[3] += fv[j] * ws.w;
    pd[0] += fv[j] * wd.x; pd[1] += fv[j] * wd.y;
    pd[2] += fv[j] * wd.z; pd[3] += fv[j] * wd.w;
  }
#pragma unroll
  for (int off = 16; off >= 1; off >>= 1) {
#pragma unroll
    for (int h = 0; h < NHEAD; ++h) {
      ps[h] += __shfl_xor(ps[h], off);
      pd[h] += __shfl_xor(pd[h], off);
    }
  }
  if (lane == 0) {
#pragma unroll
    for (int h = 0; h < NHEAD; ++h) {
      aS[(size_t)node * NHEAD + h] = ps[h];
      aD[(size_t)node * NHEAD + h] = pd[h];
    }
  }
}

// ---------------------------------------------------------------------------
// Kernel 5: edge logits (leaky relu) + segment max via sortable-uint atomicMax
// ---------------------------------------------------------------------------
__global__ __launch_bounds__(256) void logits_kernel(
    const float* __restrict__ aS, const float* __restrict__ aD,
    const int* __restrict__ src, const int* __restrict__ dst,
    float* __restrict__ logit, unsigned* __restrict__ emax, int e_cnt) {
  int e = blockIdx.x * blockDim.x + threadIdx.x;
  if (e >= e_cnt) return;
  int s = src[e], d = dst[e];
  float4 as = *(const float4*)(aS + (size_t)s * NHEAD);
  float4 ad = *(const float4*)(aD + (size_t)d * NHEAD);
  float v[NHEAD] = {as.x + ad.x, as.y + ad.y, as.z + ad.z, as.w + ad.w};
#pragma unroll
  for (int h = 0; h < NHEAD; ++h) {
    float x = (v[h] > 0.0f) ? v[h] : 0.2f * v[h];
    logit[(size_t)e * NHEAD + h] = x;
    atomicMax(emax + (size_t)d * NHEAD + h, enc_f(x));
  }
}

// ---------------------------------------------------------------------------
// Kernel 6: exp(e - max) in place + denominator atomicAdd
// ---------------------------------------------------------------------------
__global__ __launch_bounds__(256) void expsum_kernel(
    const int* __restrict__ dst, float* __restrict__ logit,
    const unsigned* __restrict__ emax, float* __restrict__ denom, int e_cnt) {
  int e = blockIdx.x * blockDim.x + threadIdx.x;
  if (e >= e_cnt) return;
  int d = dst[e];
#pragma unroll
  for (int h = 0; h < NHEAD; ++h) {
    float mx = dec_f(emax[(size_t)d * NHEAD + h]);
    float ex = __expf(logit[(size_t)e * NHEAD + h] - mx);
    logit[(size_t)e * NHEAD + h] = ex;
    atomicAdd(denom + (size_t)d * NHEAD + h, ex);
  }
}

// ---------------------------------------------------------------------------
// Kernel 7: aggregation. One wave per chunk of EC=16 consecutive edges
// (dst is sorted). Lane owns 8 of 256 feature cols (head h = lane>>3).
// Register-accumulate while dst unchanged; flush via f32 atomics.
// ---------------------------------------------------------------------------
#define EC 16
__global__ __launch_bounds__(256) void aggregate_kernel(
    const float* __restrict__ hsrc, const float* __restrict__ ex,
    const float* __restrict__ denom, const int* __restrict__ src,
    const int* __restrict__ dst, float* __restrict__ out, int e_cnt) {
  int wave = blockIdx.x * (blockDim.x >> 5) + (threadIdx.x >> 5);
  int lane = threadIdx.x & 31;
  long e0 = (long)wave * EC;
  if (e0 >= e_cnt) return;
  long e1 = e0 + EC;
  if (e1 > e_cnt) e1 = e_cnt;

  const int col = lane * 8;
  const int h   = lane >> 3;
  float acc[8] = {0, 0, 0, 0, 0, 0, 0, 0};
  int cur = dst[e0];

  for (long e = e0; e < e1; ++e) {
    int d = dst[e];
    if (d != cur) {
      float* op = out + (size_t)cur * D_IN + col;
#pragma unroll
      for (int j = 0; j < 8; ++j) {
        atomicAdd(op + j, acc[j]);
        acc[j] = 0.0f;
      }
      cur = d;
    }
    int s = src[e];
    float a = ex[(size_t)e * NHEAD + h] / denom[(size_t)d * NHEAD + h];
    const float4* hp = (const float4*)(hsrc + (size_t)s * D_IN + col);
    float4 p0 = hp[0], p1 = hp[1];
    acc[0] += p0.x * a; acc[1] += p0.y * a;
    acc[2] += p0.z * a; acc[3] += p0.w * a;
    acc[4] += p1.x * a; acc[5] += p1.y * a;
    acc[6] += p1.z * a; acc[7] += p1.w * a;
    if (e + 1 < e1)  // pull next gather row toward L0/L2 (global_prefetch_b8)
      __builtin_prefetch(hsrc + (size_t)src[e + 1] * D_IN + col, 0, 3);
  }
  float* op = out + (size_t)cur * D_IN + col;
#pragma unroll
  for (int j = 0; j < 8; ++j) atomicAdd(op + j, acc[j]);
}

// ---------------------------------------------------------------------------
extern "C" void kernel_launch(void* const* d_in, const int* in_sizes, int n_in,
                              void* d_out, int out_size, void* d_ws,
                              size_t ws_size, hipStream_t stream) {
  const float* feat  = (const float*)d_in[0];
  const float* W_src = (const float*)d_in[1];
  const float* W_dst = (const float*)d_in[2];
  const float* b_dst = (const float*)d_in[3];
  const float* WaS   = (const float*)d_in[4];
  const float* WaD   = (const float*)d_in[5];
  const int*   src   = (const int*)d_in[6];
  const int*   dst   = (const int*)d_in[7];
  float* out = (float*)d_out;

  const int N = in_sizes[0] / D_IN;   // 100000
  const int E = in_sizes[6];          // 1600000

  char* ws = (char*)d_ws;
  float*    hsrc  = (float*)(ws + OFF_HSRC);
  float*    logit = (float*)(ws + OFF_LOGIT);
  unsigned* emax  = (unsigned*)(ws + OFF_EMAX);
  float*    denom = (float*)(ws + OFF_DENOM);
  __bf16*   wpack = (__bf16*)(ws + OFF_WPACK);

  // attention a_src/a_dst live after wpack (100000*4*4 each, 32B aligned)
  float* aS = (float*)(ws + OFF_WPACK + 262144);
  float* aD = aS + (size_t)N * NHEAD;

  // 1) pack weights to bf16 WMMA B-fragments
  pack_w_kernel<<<(2 * 8 * 16 * 32 + 255) / 256, 256, 0, stream>>>(
      W_src, W_dst, wpack);
  // 2) reset softmax reduction buffers (required every call)
  init_kernel<<<(N * NHEAD + 255) / 256, 256, 0, stream>>>(emax, denom,
                                                           N * NHEAD);
  // 3) dense projections: h_src -> ws, h_dst + bias -> d_out (residual init)
  gemm_kernel<<<N / 16, 128, 0, stream>>>(feat, wpack, b_dst, hsrc, out);
  // 4) attention projections
  attn_proj_kernel<<<(N + 7) / 8, 256, 0, stream>>>(feat, WaS, WaD, aS, aD, N);
  // 5) edge logits + segment max
  logits_kernel<<<(E + 255) / 256, 256, 0, stream>>>(aS, aD, src, dst, logit,
                                                     emax, E);
  // 6) exp + denominator
  expsum_kernel<<<(E + 255) / 256, 256, 0, stream>>>(dst, logit, emax, denom,
                                                     E);
  // 7) normalized message aggregation into residual-initialized out
  int waves = (E + EC - 1) / EC;
  aggregate_kernel<<<(waves + 7) / 8, 256, 0, stream>>>(hsrc, logit, denom,
                                                        src, dst, out, E);
}
